// CrossAttentionLayer_70738111365466
// MI455X (gfx1250) — matrix operand order repaired and checked by reading, not compiled, not run on hardware
//
#include <hip/hip_runtime.h>
#include <cstdint>

#define DIM   1024
#define SEQ   2048
#define BATCH 4
#define MTOT  (BATCH * SEQ)

typedef __bf16 bf16;
typedef __attribute__((ext_vector_type(16))) __bf16 v16bf;
typedef __attribute__((ext_vector_type(8)))  __bf16 v8bf;
typedef __attribute__((ext_vector_type(8)))  float  v8f;
typedef __attribute__((ext_vector_type(4)))  unsigned int u32x4;
typedef __attribute__((ext_vector_type(8)))  int i32x8;
typedef __attribute__((ext_vector_type(4)))  int i32x4;

__device__ __forceinline__ v8f wmma_bf16(v16bf a, v16bf b, v8f c) {
  return __builtin_amdgcn_wmma_f32_16x16x32_bf16(false, a, false, b, (short)0, c,
                                                 false, false);
}

// ---------------------------------------------------------------------------
// Tensor Data Mover: DMA a 2D bf16 tile (tile_d1 rows x tile_d0 elems, row
// stride stride0 elems) from global memory into LDS (rows packed contiguously).
// D# bitfields per CDNA5 ISA ch.8. Wave-level op, EXEC-ignored; TENSORcnt.
// ---------------------------------------------------------------------------
__device__ __forceinline__ void tdm_load_2d(unsigned int lds_byte_off,
                                            const bf16* gptr,
                                            unsigned int tensor_d0,
                                            unsigned int tensor_d1,
                                            unsigned int tile_d0,
                                            unsigned int tile_d1,
                                            unsigned long long stride0) {
  unsigned long long ga = (unsigned long long)(uintptr_t)gptr;
  u32x4 g0;
  g0[0] = 1u;                                    // count=1, user descriptor
  g0[1] = lds_byte_off;                          // lds_addr [63:32]
  g0[2] = (unsigned int)ga;                      // global_addr [95:64]
  g0[3] = (unsigned int)(ga >> 32) | (2u << 30); // global_addr [120:96] | type=2
  i32x8 g1;
  g1[0] = (int)(1u << 16);                       // data_size=1 (2 bytes/elem)
  g1[1] = (int)((tensor_d0 & 0xFFFFu) << 16);    // tensor_dim0[15:0]
  g1[2] = (int)((tensor_d0 >> 16) | ((tensor_d1 & 0xFFFFu) << 16));
  g1[3] = (int)((tensor_d1 >> 16) | ((tile_d0 & 0xFFFFu) << 16));
  g1[4] = (int)(tile_d1 & 0xFFFFu);              // tile_dim1; tile_dim2=0
  g1[5] = (int)(unsigned int)(stride0 & 0xFFFFFFFFu);       // dim0 stride lo
  g1[6] = (int)(unsigned int)((stride0 >> 32) & 0xFFFFu);   // stride hi; dim1 stride=0
  g1[7] = 0;
  i32x4 zg = {0, 0, 0, 0};
#if defined(__clang_major__) && (__clang_major__ >= 23)
  i32x8 zg8 = {0, 0, 0, 0, 0, 0, 0, 0};
  __builtin_amdgcn_tensor_load_to_lds(g0, g1, zg, zg, zg8, 0);
#else
  __builtin_amdgcn_tensor_load_to_lds(g0, g1, zg, zg, 0);
#endif
}

__device__ __forceinline__ unsigned int lds_offset_of(const void* p) {
  // Generic LDS addresses carry the LDS byte offset in bits [31:0] (ISA 10.2).
  return (unsigned int)(uintptr_t)p;
}

// A fragment: 16x32 (MxK) bf16 tile, source row-major with leading dim ld.
// Lane L (L<16) = row L, holds K {0..15}; lane L+16 holds K {16..31}.
__device__ __forceinline__ v16bf load_frag_a(const bf16* p, int ld, int lane) {
  const bf16* r = p + (size_t)(lane & 15) * ld + ((lane >> 4) * 16);
  v8bf lo = *(const v8bf*)r;
  v8bf hi = *(const v8bf*)(r + 8);
  v16bf f;
#pragma unroll
  for (int j = 0; j < 8; ++j) { f[j] = lo[j]; f[j + 8] = hi[j]; }
  return f;
}

// B fragment: 32x16 (KxN) bf16 tile where memory holds rows of N with K
// contiguous (row-major [N][K], leading dim ld). Lane = column, halves split K.
__device__ __forceinline__ v16bf load_frag_b(const bf16* p, int ld, int lane) {
  return *(const v16bf*)(p + (size_t)(lane & 15) * ld + (lane >> 4) * 16);
}

// ---------------------------------------------------------------------------
// Kernel 1: f32 -> bf16 conversion (x and weights)
// ---------------------------------------------------------------------------
__global__ void ca_cvt_bf16(const float* __restrict__ src, bf16* __restrict__ dst,
                            int n) {
  int i = (blockIdx.x * blockDim.x + threadIdx.x) * 4;
  if (i + 3 < n) {
    float4 v = *(const float4*)(src + i);
    dst[i + 0] = (bf16)v.x;
    dst[i + 1] = (bf16)v.y;
    dst[i + 2] = (bf16)v.z;
    dst[i + 3] = (bf16)v.w;
  }
}

// ---------------------------------------------------------------------------
// Kernel 2: QKV projection  out[m,n] = sum_k x[m,k]*W[n,k] + bias[n]
// Workgroup: 128 threads (4 waves), tile 128(M) x 64(N); wave tile 64x32.
// Double-buffered LDS tiles filled by the Tensor Data Mover: wave 0 issues the
// next tile's 2 DMAs, then waits tensorcnt<=2 (in-order completion guarantees
// the current tile's DMAs have landed while the new pair stays in flight).
// ---------------------------------------------------------------------------
__global__ void ca_qkv_gemm(const bf16* __restrict__ xb,
                            const bf16* __restrict__ wq, const bf16* __restrict__ wk,
                            const bf16* __restrict__ wv,
                            const float* __restrict__ bq, const float* __restrict__ bk,
                            const float* __restrict__ bv,
                            bf16* __restrict__ qb, bf16* __restrict__ kb,
                            bf16* __restrict__ vb) {
  const bf16* W;
  const float* bias;
  bf16* out;
  if (blockIdx.z == 0)      { W = wq; bias = bq; out = qb; }
  else if (blockIdx.z == 1) { W = wk; bias = bk; out = kb; }
  else                      { W = wv; bias = bv; out = vb; }

  __shared__ __align__(16) bf16 As[2][128 * 32];
  __shared__ __align__(16) bf16 Bs[2][64 * 32];

  const int tid = threadIdx.x;
  const int lane = tid & 31, w = tid >> 5;
  const int m0 = blockIdx.x * 128, n0 = blockIdx.y * 64;
  const int wm = (w >> 1) * 64;
  const int wn = (w & 1) * 32;

  const bf16* arow = xb + (size_t)m0 * DIM;
  const bf16* brow = W + (size_t)n0 * DIM;

  if (w == 0) {  // prologue: fill buffer 0
    tdm_load_2d(lds_offset_of(As[0]), arow, DIM, MTOT, 32, 128, DIM);
    tdm_load_2d(lds_offset_of(Bs[0]), brow, DIM, DIM, 32, 64, DIM);
  }

  v8f acc[4][2] = {};
  for (int k0 = 0; k0 < DIM; k0 += 32) {
    const int buf = (k0 >> 5) & 1;
    if (w == 0) {
      if (k0 + 32 < DIM) {  // kick off next tile before waiting on current
        tdm_load_2d(lds_offset_of(As[buf ^ 1]), arow + k0 + 32, DIM, MTOT, 32, 128, DIM);
        tdm_load_2d(lds_offset_of(Bs[buf ^ 1]), brow + k0 + 32, DIM, DIM, 32, 64, DIM);
        __builtin_amdgcn_s_wait_tensorcnt(2);
      } else {
        __builtin_amdgcn_s_wait_tensorcnt(0);
      }
    }
    __syncthreads();
    v16bf bfr[2];
#pragma unroll
    for (int nt = 0; nt < 2; ++nt)
      bfr[nt] = load_frag_b(Bs[buf] + (wn + nt * 16) * 32, 32, lane);
#pragma unroll
    for (int mt = 0; mt < 4; ++mt) {
      v16bf afr = load_frag_a(As[buf] + (wm + mt * 16) * 32, 32, lane);
#pragma unroll
      for (int nt = 0; nt < 2; ++nt)
        acc[mt][nt] = wmma_bf16(afr, bfr[nt], acc[mt][nt]);
    }
    __syncthreads();
  }
#pragma unroll
  for (int mt = 0; mt < 4; ++mt)
#pragma unroll
    for (int nt = 0; nt < 2; ++nt)
#pragma unroll
      for (int r = 0; r < 8; ++r) {
        int row = m0 + wm + mt * 16 + (lane >> 4) * 8 + r;
        int col = n0 + wn + nt * 16 + (lane & 15);
        out[(size_t)row * DIM + col] = (bf16)(acc[mt][nt][r] + bias[col]);
      }
}

// ---------------------------------------------------------------------------
// Kernel 3: scores = (q @ k^T) * scale, row softmax -> attn (bf16)
// Workgroup: 256 threads (8 waves) per (batch, 16 query rows).
// q tile (16x1024) staged via TDM; wave w owns key columns [w*256, w*256+256).
// ---------------------------------------------------------------------------
__global__ void ca_scores_softmax(const bf16* __restrict__ qb,
                                  const bf16* __restrict__ kb,
                                  bf16* __restrict__ attn) {
  const int b = blockIdx.y;
  const int q0 = blockIdx.x * 16;
  const int tid = threadIdx.x, lane = tid & 31, w = tid >> 5;

  __shared__ __align__(16) bf16 qs[16 * DIM];
  __shared__ float red[8][16];

  if (w == 0) {  // stage 16x1024 q tile with the TDM
    tdm_load_2d(lds_offset_of(qs), qb + (size_t)b * SEQ * DIM + (size_t)q0 * DIM,
                DIM, SEQ, DIM, 16, DIM);
    __builtin_amdgcn_s_wait_tensorcnt(0);
  }
  __syncthreads();

  const int c0 = w * 256;
  const bf16* kbase = kb + (size_t)b * SEQ * DIM;
  v8f acc[16] = {};
  for (int k0 = 0; k0 < DIM; k0 += 32) {
    v16bf afr = load_frag_a(qs + k0, DIM, lane);
#pragma unroll
    for (int nt = 0; nt < 16; ++nt) {
      v16bf bfr = load_frag_b(kbase + (size_t)(c0 + nt * 16) * DIM + k0, DIM, lane);
      acc[nt] = wmma_bf16(afr, bfr, acc[nt]);
    }
  }

  const float scale = 0.03125f;  // 1/sqrt(1024)
  // --- row max ---
  float pm[8];
#pragma unroll
  for (int r = 0; r < 8; ++r) {
    float m = -3.0e38f;
#pragma unroll
    for (int nt = 0; nt < 16; ++nt) {
      acc[nt][r] *= scale;
      m = fmaxf(m, acc[nt][r]);
    }
#pragma unroll
    for (int d = 1; d < 16; d <<= 1) m = fmaxf(m, __shfl_xor(m, d, 32));
    pm[r] = m;
  }
  if ((lane & 15) == 0)
#pragma unroll
    for (int r = 0; r < 8; ++r) red[w][(lane >> 4) * 8 + r] = pm[r];
  __syncthreads();
  float gm[8];
#pragma unroll
  for (int r = 0; r < 8; ++r) {
    int row = (lane >> 4) * 8 + r;
    float m = red[0][row];
#pragma unroll
    for (int w2 = 1; w2 < 8; ++w2) m = fmaxf(m, red[w2][row]);
    gm[r] = m;
  }
  __syncthreads();
  // --- exp + row sum ---
  float ps[8];
#pragma unroll
  for (int r = 0; r < 8; ++r) {
    float s = 0.f;
#pragma unroll
    for (int nt = 0; nt < 16; ++nt) {
      acc[nt][r] = __expf(acc[nt][r] - gm[r]);
      s += acc[nt][r];
    }
#pragma unroll
    for (int d = 1; d < 16; d <<= 1) s += __shfl_xor(s, d, 32);
    ps[r] = s;
  }
  if ((lane & 15) == 0)
#pragma unroll
    for (int r = 0; r < 8; ++r) red[w][(lane >> 4) * 8 + r] = ps[r];
  __syncthreads();
  bf16* abase = attn + (size_t)b * SEQ * SEQ;
#pragma unroll
  for (int r = 0; r < 8; ++r) {
    int row = (lane >> 4) * 8 + r;
    float s = 0.f;
#pragma unroll
    for (int w2 = 0; w2 < 8; ++w2) s += red[w2][row];
    float inv = 1.f / s;
#pragma unroll
    for (int nt = 0; nt < 16; ++nt)
      abase[(size_t)(q0 + row) * SEQ + c0 + nt * 16 + (lane & 15)] =
          (bf16)(acc[nt][r] * inv);
  }
}

// ---------------------------------------------------------------------------
// Kernel 4: out_vision = attn @ v   (A direct from global, V transposed via
// LDS). Global tile loads software-pipelined one iteration ahead so the VMEM
// latency overlaps the WMMA stream.
// ---------------------------------------------------------------------------
__global__ void ca_attn_v(const bf16* __restrict__ attn, const bf16* __restrict__ vb,
                          float* __restrict__ outv) {
  const int b = blockIdx.z, m0 = blockIdx.x * 128, n0 = blockIdx.y * 128;
  const int tid = threadIdx.x, lane = tid & 31, w = tid >> 5;
  const int wm = (w >> 2) * 64, wn = (w & 3) * 32;
  __shared__ __align__(16) bf16 Vt[128 * 32];  // [n][k]

  const bf16* ab = attn + (size_t)b * SEQ * SEQ;
  const bf16* vbb = vb + (size_t)b * SEQ * DIM;
  v8f acc[4][2] = {};

  const int kl = tid >> 3;         // 0..31
  const int nc = (tid & 7) * 16;   // 0..112
  v16bf t = *(const v16bf*)(vbb + (size_t)kl * DIM + n0 + nc);  // prologue
  for (int k0 = 0; k0 < SEQ; k0 += 32) {
#pragma unroll
    for (int j = 0; j < 16; ++j) Vt[(nc + j) * 32 + kl] = t[j];
    __syncthreads();
    if (k0 + 32 < SEQ)  // prefetch next tile into registers during compute
      t = *(const v16bf*)(vbb + (size_t)(k0 + 32 + kl) * DIM + n0 + nc);
    v16bf bfr[2];
#pragma unroll
    for (int nt = 0; nt < 2; ++nt)
      bfr[nt] = load_frag_b(Vt + (wn + nt * 16) * 32, 32, lane);
#pragma unroll
    for (int mt = 0; mt < 4; ++mt) {
      v16bf afr = load_frag_a(ab + (size_t)(m0 + wm + mt * 16) * SEQ + k0, SEQ, lane);
#pragma unroll
      for (int nt = 0; nt < 2; ++nt)
        acc[mt][nt] = wmma_bf16(afr, bfr[nt], acc[mt][nt]);
    }
    __syncthreads();
  }
  float* ob = outv + (size_t)b * SEQ * DIM;
#pragma unroll
  for (int mt = 0; mt < 4; ++mt)
#pragma unroll
    for (int nt = 0; nt < 2; ++nt)
#pragma unroll
      for (int r = 0; r < 8; ++r) {
        int row = m0 + wm + mt * 16 + (lane >> 4) * 8 + r;
        int col = n0 + wn + nt * 16 + (lane & 15);
        ob[(size_t)row * DIM + col] = acc[mt][nt][r];
      }
}

// ---------------------------------------------------------------------------
// Kernel 5: out_text = attn^T @ x   (both operands transposed via LDS),
// global tile loads software-pipelined one iteration ahead.
// ---------------------------------------------------------------------------
__global__ void ca_attn_t_x(const bf16* __restrict__ attn, const bf16* __restrict__ xb,
                            float* __restrict__ outt) {
  const int b = blockIdx.z, m0 = blockIdx.x * 128, n0 = blockIdx.y * 128;
  const int tid = threadIdx.x, lane = tid & 31, w = tid >> 5;
  const int wm = (w >> 2) * 64, wn = (w & 3) * 32;
  __shared__ __align__(16) bf16 At[128 * 32];  // [kpos][q]
  __shared__ __align__(16) bf16 Xt[128 * 32];  // [d][q]

  const bf16* ab = attn + (size_t)b * SEQ * SEQ;
  const bf16* xbb = xb + (size_t)b * SEQ * DIM;
  v8f acc[4][2] = {};

  const int ql = tid >> 3;        // 0..31
  const int c = (tid & 7) * 16;   // 0..112
  v16bf ta = *(const v16bf*)(ab + (size_t)ql * SEQ + m0 + c);    // prologue
  v16bf tx = *(const v16bf*)(xbb + (size_t)ql * DIM + n0 + c);
  for (int q0 = 0; q0 < SEQ; q0 += 32) {
#pragma unroll
    for (int j = 0; j < 16; ++j) {
      At[(c + j) * 32 + ql] = ta[j];
      Xt[(c + j) * 32 + ql] = tx[j];
    }
    __syncthreads();
    if (q0 + 32 < SEQ) {  // prefetch next tiles into registers during compute
      ta = *(const v16bf*)(ab + (size_t)(q0 + 32 + ql) * SEQ + m0 + c);
      tx = *(const v16bf*)(xbb + (size_t)(q0 + 32 + ql) * DIM + n0 + c);
    }
    v16bf bfr[2];
#pragma unroll
    for (int nt = 0; nt < 2; ++nt)
      bfr[nt] = load_frag_b(Xt + (wn + nt * 16) * 32, 32, lane);
#pragma unroll
    for (int mt = 0; mt < 4; ++mt) {
      v16bf afr = load_frag_a(At + (wm + mt * 16) * 32, 32, lane);
#pragma unroll
      for (int nt = 0; nt < 2; ++nt)
        acc[mt][nt] = wmma_bf16(afr, bfr[nt], acc[mt][nt]);
    }
    __syncthreads();
  }
  float* ob = outt + (size_t)b * SEQ * DIM;
#pragma unroll
  for (int mt = 0; mt < 4; ++mt)
#pragma unroll
    for (int nt = 0; nt < 2; ++nt)
#pragma unroll
      for (int r = 0; r < 8; ++r) {
        int row = m0 + wm + mt * 16 + (lane >> 4) * 8 + r;
        int col = n0 + wn + nt * 16 + (lane & 15);
        ob[(size_t)row * DIM + col] = acc[mt][nt][r];
      }
}

// ---------------------------------------------------------------------------
extern "C" void kernel_launch(void* const* d_in, const int* in_sizes, int n_in,
                              void* d_out, int out_size, void* d_ws, size_t ws_size,
                              hipStream_t stream) {
  (void)in_sizes; (void)n_in; (void)out_size; (void)ws_size;
  const float* x  = (const float*)d_in[0];
  const float* Wq = (const float*)d_in[1];
  const float* bq = (const float*)d_in[2];
  const float* Wk = (const float*)d_in[3];
  const float* bk = (const float*)d_in[4];
  const float* Wv = (const float*)d_in[5];
  const float* bv = (const float*)d_in[6];
  float* out = (float*)d_out;

  bf16* ws  = (bf16*)d_ws;
  bf16* xb  = ws;                                   // 8,388,608 elems
  bf16* wqb = xb + (size_t)MTOT * DIM;
  bf16* wkb = wqb + (size_t)DIM * DIM;
  bf16* wvb = wkb + (size_t)DIM * DIM;
  bf16* qb  = wvb + (size_t)DIM * DIM;
  bf16* kb  = qb + (size_t)MTOT * DIM;
  bf16* vb  = kb + (size_t)MTOT * DIM;
  bf16* attn = vb + (size_t)MTOT * DIM;             // 16,777,216 elems

  ca_cvt_bf16<<<(MTOT * DIM) / 4 / 256, 256, 0, stream>>>(x, xb, MTOT * DIM);
  ca_cvt_bf16<<<(DIM * DIM) / 4 / 256, 256, 0, stream>>>(Wq, wqb, DIM * DIM);
  ca_cvt_bf16<<<(DIM * DIM) / 4 / 256, 256, 0, stream>>>(Wk, wkb, DIM * DIM);
  ca_cvt_bf16<<<(DIM * DIM) / 4 / 256, 256, 0, stream>>>(Wv, wvb, DIM * DIM);

  ca_qkv_gemm<<<dim3(MTOT / 128, DIM / 64, 3), 128, 0, stream>>>(
      xb, wqb, wkb, wvb, bq, bk, bv, qb, kb, vb);

  ca_scores_softmax<<<dim3(SEQ / 16, BATCH), 256, 0, stream>>>(qb, kb, attn);

  ca_attn_v<<<dim3(SEQ / 128, DIM / 128, BATCH), 256, 0, stream>>>(attn, vb, out);

  ca_attn_t_x<<<dim3(SEQ / 128, DIM / 128, BATCH), 256, 0, stream>>>(
      attn, xb, out + (size_t)BATCH * SEQ * DIM);
}
